// HeadStagePsg_72361609003195
// MI455X (gfx1250) — matrix operand order repaired and checked by reading, not compiled, not run on hardware
//
#include <hip/hip_runtime.h>
#include <math.h>

// ---------------------------------------------------------------------------
// MI455X (gfx1250) implementation of the 1-D Swin encoder/decoder reference.
// All dense GEMMs run through v_wmma_f32_16x16x32_bf16 (fp32 -> bf16 inline,
// fp32 accumulate), 4 N-tiles per wave. sched_group_barrier forces the
// pipeline per k-step: 20x global_load_b128 batch -> 40x v_cvt_pk_bf16_f32
// -> 4x v_wmma, so loads overlap and WMMA WAR hazards are buried.
// Attention (ws<=45, hd=32, <10% of FLOPs) is a VALU kernel with k/v staged
// in LDS per window. Model state (<200MB) is L2-resident on MI455X (192MB L2,
// 23.3TB/s HBM), so the GEMMs are compute-bound -> WMMA is the right path.
// ---------------------------------------------------------------------------

typedef __attribute__((ext_vector_type(16))) __bf16 v16bf;
typedef __attribute__((ext_vector_type(8)))  float  v8f;

__device__ __forceinline__ void cvt8(v16bf& v, int base, float4 a, float4 b) {
  v[base + 0] = (__bf16)a.x; v[base + 1] = (__bf16)a.y;
  v[base + 2] = (__bf16)a.z; v[base + 3] = (__bf16)a.w;
  v[base + 4] = (__bf16)b.x; v[base + 5] = (__bf16)b.y;
  v[base + 6] = (__bf16)b.z; v[base + 7] = (__bf16)b.w;
}

// ============================ generic WMMA GEMM ============================
// Y[M,N] = act(X[M,K] @ W[N,K]^T + bias) + res
// One wave computes a 16x64 output tile (1 M-tile x 4 N-tiles); 8 waves/block.
// Requirements (guaranteed by host): K % 32 == 0, N % 64 == 0, and A/Y/res
// have at least Mtiles*16 addressable rows (workspace slack covers M=2520).
// A frag (16x32 bf16): half=lane>>4, m=lane&15; elems 0..15 <- K run at
//   kt+16*half (contiguous float4s).
// B frag (32x16 bf16): n=lane&15; elems 0..15 <- K run at kt+16*half.
// D (16x16 f32): elem r -> M = r + 8*half, N = lane&15.
__global__ __launch_bounds__(256) void gemm_bf16_wmma(
    const float* __restrict__ A, const float* __restrict__ Wt,
    const float* __restrict__ bias, const float* __restrict__ res,
    float* __restrict__ Y, int Mtiles, int N, int K, int lda, int ldw, int act)
{
  const int lane = threadIdx.x & 31;
  const int wave = threadIdx.x >> 5;
  const int ngrp = N >> 6;  // groups of 4 n-tiles
  const long total = (long)Mtiles * ngrp;
  const long tile  = (long)blockIdx.x * 8 + wave;
  if (tile >= total) return;
  // mt fastest: the 8 waves of a block share the same weight (B) stream so
  // B loads hit WGP$ after the first wave; A streams are per-wave.
  const int mt = (int)(tile % Mtiles);
  const int ng = (int)(tile / Mtiles);
  const int m0 = mt << 4, n0 = ng << 6;
  const int half = lane >> 4;
  const int mr   = lane & 15;

  const float* __restrict__ arow = A + (size_t)(m0 + mr) * lda;
  const float* __restrict__ wr0 = Wt + (size_t)(n0 + mr) * ldw;
  const float* __restrict__ wr1 = wr0 + (size_t)16 * ldw;
  const float* __restrict__ wr2 = wr1 + (size_t)16 * ldw;
  const float* __restrict__ wr3 = wr2 + (size_t)16 * ldw;

  v8f acc0 = {}, acc1 = {}, acc2 = {}, acc3 = {};

#pragma unroll 2
  for (int kt = 0; kt < K; kt += 32) {
    // ---- batch all 20 b128 loads first (independent dests -> MLP) ----
    const float* ap = arow + kt + 16 * half;
    const float4 ra0 = *(const float4*)(ap);
    const float4 ra1 = *(const float4*)(ap + 4);
    const float4 ra2 = *(const float4*)(ap + 8);
    const float4 ra3 = *(const float4*)(ap + 12);
    const float* bp0 = wr0 + kt + 16 * half;
    const float4 b00 = *(const float4*)(bp0);
    const float4 b01 = *(const float4*)(bp0 + 4);
    const float4 b02 = *(const float4*)(bp0 + 8);
    const float4 b03 = *(const float4*)(bp0 + 12);
    const float* bp1 = wr1 + kt + 16 * half;
    const float4 b10 = *(const float4*)(bp1);
    const float4 b11 = *(const float4*)(bp1 + 4);
    const float4 b12 = *(const float4*)(bp1 + 8);
    const float4 b13 = *(const float4*)(bp1 + 12);
    const float* bp2 = wr2 + kt + 16 * half;
    const float4 b20 = *(const float4*)(bp2);
    const float4 b21 = *(const float4*)(bp2 + 4);
    const float4 b22 = *(const float4*)(bp2 + 8);
    const float4 b23 = *(const float4*)(bp2 + 12);
    const float* bp3 = wr3 + kt + 16 * half;
    const float4 b30 = *(const float4*)(bp3);
    const float4 b31 = *(const float4*)(bp3 + 4);
    const float4 b32 = *(const float4*)(bp3 + 8);
    const float4 b33 = *(const float4*)(bp3 + 12);

    // ---- convert everything ----
    v16bf a, f0, f1, f2, f3;
    cvt8(a, 0, ra0, ra1); cvt8(a, 8, ra2, ra3);
    cvt8(f0, 0, b00, b01); cvt8(f0, 8, b02, b03);
    cvt8(f1, 0, b10, b11); cvt8(f1, 8, b12, b13);
    cvt8(f2, 0, b20, b21); cvt8(f2, 8, b22, b23);
    cvt8(f3, 0, b30, b31); cvt8(f3, 8, b32, b33);

    // ---- 4 independent WMMAs back-to-back ----
    acc0 = __builtin_amdgcn_wmma_f32_16x16x32_bf16(
        false, a, false, f0, (short)0, acc0, false, false);
    acc1 = __builtin_amdgcn_wmma_f32_16x16x32_bf16(
        false, a, false, f1, (short)0, acc1, false, false);
    acc2 = __builtin_amdgcn_wmma_f32_16x16x32_bf16(
        false, a, false, f2, (short)0, acc2, false, false);
    acc3 = __builtin_amdgcn_wmma_f32_16x16x32_bf16(
        false, a, false, f3, (short)0, acc3, false, false);

    // ---- force pipeline stages: VMEM-read batch -> VALU cvt -> WMMA ----
    __builtin_amdgcn_sched_group_barrier(0x020, 20, 0); // 20 vmem reads
    __builtin_amdgcn_sched_group_barrier(0x002, 64, 0); // cvt + addr valu
    __builtin_amdgcn_sched_group_barrier(0x008, 4, 0);  // 4 wmma
  }

  v8f accs[4] = {acc0, acc1, acc2, acc3};
#pragma unroll
  for (int t = 0; t < 4; t++) {
    const int n = n0 + 16 * t + mr;
#pragma unroll
    for (int r = 0; r < 8; r++) {
      const int mrow = m0 + r + 8 * half;
      float v = accs[t][r];
      if (bias) v += bias[n];
      if (act)  v = 0.5f * v * (1.0f + erff(v * 0.70710678118654752f)); // GELU
      if (res)  v += res[(size_t)mrow * N + n];
      Y[(size_t)mrow * N + n] = v;
    }
  }
}

// ============================== LayerNorm ==================================
// One block (128 thr) per row; C in {128, 768, 512}. Optional broadcast add
// (ape, indexed row%Lmod) and residual add epilogues.
__global__ __launch_bounds__(128) void ln_kernel(
    const float* __restrict__ X, float* __restrict__ Y,
    const float* __restrict__ g, const float* __restrict__ b,
    int C, const float* __restrict__ addB, int Lmod,
    const float* __restrict__ addR)
{
  __shared__ float red[128];
  const int row = blockIdx.x;
  const float* __restrict__ xr = X + (size_t)row * C;
  float sum = 0.f;
  for (int c = threadIdx.x; c < C; c += 128) sum += xr[c];
  red[threadIdx.x] = sum; __syncthreads();
  for (int s = 64; s > 0; s >>= 1) {
    if (threadIdx.x < s) red[threadIdx.x] += red[threadIdx.x + s];
    __syncthreads();
  }
  const float mean = red[0] / (float)C;
  __syncthreads();
  float vs = 0.f;
  for (int c = threadIdx.x; c < C; c += 128) {
    float d = xr[c] - mean; vs += d * d;
  }
  red[threadIdx.x] = vs; __syncthreads();
  for (int s = 64; s > 0; s >>= 1) {
    if (threadIdx.x < s) red[threadIdx.x] += red[threadIdx.x + s];
    __syncthreads();
  }
  const float rinv = rsqrtf(red[0] / (float)C + 1e-5f);
  float* __restrict__ yr = Y + (size_t)row * C;
  const float* __restrict__ ar = addB ? (addB + (size_t)(row % Lmod) * C) : nullptr;
  const float* __restrict__ rr = addR ? (addR + (size_t)row * C) : nullptr;
  for (int c = threadIdx.x; c < C; c += 128) {
    float v = (xr[c] - mean) * rinv * g[c] + b[c];
    if (ar) v += ar[c];
    if (rr) v += rr[c];
    yr[c] = v;
  }
}

// ========================= windowed attention ==============================
// One block handles WPB windows. k/v for the block's windows staged in LDS.
// Thread = (window slot, token i, head). Roll-shift folded into index math:
// rolled pos r <-> global g = (r + shift) mod L (read qkv and write o at g).
template <int WS, int SHIFT_NZ>
__global__ __launch_bounds__(256) void attn_kernel(
    const float* __restrict__ qkv, float* __restrict__ o,
    const float* __restrict__ rpb, int L, int shift, int nWtotal)
{
  constexpr int NH = 4, HD = 32, C = 128;
  constexpr int WPB = (WS == 6) ? 8 : 1;
  __shared__ float kv[WPB * WS * 2 * C];
  const int nW = L / WS;
  const long wbase = (long)blockIdx.x * WPB;

  // cooperative stage of k and v for the block's windows
  const int totalElems = WPB * WS * 2 * C;
  for (int e = threadIdx.x; e < totalElems; e += 256) {
    const int wslot = e / (WS * 2 * C);
    const int rem   = e % (WS * 2 * C);
    const long gw   = wbase + wslot;
    float val = 0.f;
    if (gw < nWtotal) {
      const int j  = rem / (2 * C);
      const int c2 = rem % (2 * C);
      const int bb   = (int)(gw / nW);
      const int wloc = (int)(gw % nW);
      int gg = wloc * WS + j + shift; if (gg >= L) gg -= L;
      const int col = (c2 < C) ? (C + c2) : (2 * C + (c2 - C)); // k then v
      val = qkv[((size_t)(bb * L + gg)) * 384 + col];
    }
    kv[e] = val;
  }
  __syncthreads();

  const int tid   = threadIdx.x;
  const int wslot = tid / (WS * NH);
  const int rem   = tid % (WS * NH);
  if (wslot >= WPB) return;
  const long gw = wbase + wslot;
  if (gw >= nWtotal) return;
  const int i    = rem / NH;
  const int head = rem % NH;
  const int bb   = (int)(gw / nW);
  const int wloc = (int)(gw % nW);
  const int ri = wloc * WS + i;
  int gi = ri + shift; if (gi >= L) gi -= L;

  const float* __restrict__ qp = qkv + ((size_t)(bb * L + gi)) * 384 + head * HD;
  float q[HD];
#pragma unroll
  for (int d = 0; d < HD; d++) q[d] = qp[d] * 0.17677669529663687f; // hd^-0.5

  const float* __restrict__ kbase = kv + wslot * WS * 2 * C;
  int segi = 0;
  if (SHIFT_NZ) segi = (ri >= L - WS) ? ((ri >= L - shift) ? 2 : 1) : 0;

  float s[WS];
  float mx = -1e30f;
#pragma unroll
  for (int j = 0; j < WS; j++) {
    const float* __restrict__ kj = kbase + j * 2 * C + head * HD;
    float acc = 0.f;
#pragma unroll
    for (int d = 0; d < HD; d++) acc += q[d] * kj[d];
    acc += rpb[(i - j + WS - 1) * NH + head];
    if (SHIFT_NZ) {
      const int rj = wloc * WS + j;
      const int segj = (rj >= L - WS) ? ((rj >= L - shift) ? 2 : 1) : 0;
      if (segj != segi) acc -= 100.0f;
    }
    s[j] = acc;
    mx = fmaxf(mx, acc);
  }
  float sum = 0.f;
#pragma unroll
  for (int j = 0; j < WS; j++) { s[j] = __expf(s[j] - mx); sum += s[j]; }
  const float inv = 1.0f / sum;

  float out[HD];
#pragma unroll
  for (int d = 0; d < HD; d++) out[d] = 0.f;
#pragma unroll
  for (int j = 0; j < WS; j++) {
    const float* __restrict__ vj = kbase + j * 2 * C + C + head * HD;
    const float w = s[j] * inv;
#pragma unroll
    for (int d = 0; d < HD; d++) out[d] += w * vj[d];
  }
  float* __restrict__ op = o + ((size_t)(bb * L + gi)) * C + head * HD;
#pragma unroll
  for (int d = 0; d < HD; d++) op[d] = out[d];
}

// ====================== patch-embed input staging ==========================
// Build Xpe[token, c*5+k] (K padded 540 -> 544 with zeros):
// Xpe[t, c*5+k] = x[b, l/30, l%30, c], l = 5t+k (B=8, P=1260).
__global__ void stage_pe_kernel(const float* __restrict__ x,
                                float* __restrict__ Xpe, long total)
{
  const long idx = (long)blockIdx.x * 256 + threadIdx.x;
  if (idx >= total) return;
  const int token = (int)(idx / 544);
  const int kk    = (int)(idx % 544);
  float val = 0.f;
  if (kk < 540) {
    const int c = kk / 5, k5 = kk % 5;
    const int bb = token / 7560, t = token % 7560;
    const int l = 5 * t + k5;
    const int p = l / 30, j = l % 30;
    val = x[(((size_t)bb * 1260 + p) * 30 + j) * 108 + c];
  }
  Xpe[idx] = val;
}

// Zero-pad pe_w (128 x 540) -> (128 x 544) so the GEMM K-loop is regular.
__global__ void pad_pew_kernel(const float* __restrict__ pe_w,
                               float* __restrict__ out)
{
  const int idx = blockIdx.x * 256 + threadIdx.x;
  if (idx >= 128 * 544) return;
  const int n = idx / 544, k = idx % 544;
  out[idx] = (k < 540) ? pe_w[n * 540 + k] : 0.f;
}

__global__ void copy_kernel(const float* __restrict__ s,
                            float* __restrict__ d, long n)
{
  const long i = (long)blockIdx.x * 256 + threadIdx.x;
  if (i < n) d[i] = s[i];
}

// ============================== head (N=6) =================================
__global__ void head_kernel(const float* __restrict__ h,
                            const float* __restrict__ w,
                            const float* __restrict__ bias,
                            float* __restrict__ out, int M)
{
  const int idx = blockIdx.x * 256 + threadIdx.x;
  if (idx >= M * 6) return;
  const int row = idx / 6, o = idx % 6;
  const float* __restrict__ hr = h + (size_t)row * 128;
  const float* __restrict__ wr = w + o * 128;
  float acc = 0.f;
#pragma unroll 4
  for (int c = 0; c < 128; c++) acc += hr[c] * wr[c];
  out[idx] = acc + bias[o];
}

// ============================== host side ==================================
struct Blk { const float* p[13]; };
enum { N1G = 0, N1B, QKVW, QKVB, RPB, PROJW, PROJB, N2G, N2B,
       FC1W, FC1B, FC2W, FC2B };

extern "C" void kernel_launch(void* const* d_in, const int* in_sizes, int n_in,
                              void* d_out, int out_size, void* d_ws, size_t ws_size,
                              hipStream_t stream)
{
  auto F = [&](int i) { return (const float*)d_in[i]; };

  const float *x, *pe_w, *pe_b, *pe_ng, *pe_nb, *ape;
  const float *m6_ng, *m6_nb, *m6_w, *m4_ng, *m4_nb, *m4_w;
  const float *px_w, *px_ng, *px_nb, *head_w, *head_b;
  Blk eb[2], db[4], ub[4];

  // Discriminate flattening order: insertion (in_sizes[2]==pe_w==69120) vs
  // jax-sorted pytree (in_sizes[2]==ape==967680).
  const bool sorted_order = (n_in > 2 && in_sizes[2] != 69120);
  if (!sorted_order) {
    x = F(0);
    pe_w = F(2); pe_b = F(3); pe_ng = F(4); pe_nb = F(5); ape = F(6);
    int idx = 7;
    for (int i = 0; i < 2; i++) for (int j = 0; j < 13; j++) eb[i].p[j] = F(idx++);
    m6_ng = F(idx++); m6_nb = F(idx++); m6_w = F(idx++);
    for (int i = 0; i < 4; i++) for (int j = 0; j < 13; j++) db[i].p[j] = F(idx++);
    m4_ng = F(idx++); m4_nb = F(idx++); m4_w = F(idx++);
    for (int i = 0; i < 4; i++) for (int j = 0; j < 13; j++) ub[i].p[j] = F(idx++);
    px_w = F(idx++); px_ng = F(idx++); px_nb = F(idx++);
    head_w = F(idx++); head_b = F(idx++);
  } else {
    // sorted block keys: fc1_b,fc1_w,fc2_b,fc2_w,n1b,n1g,n2b,n2g,proj_b,
    //                    proj_w,qkv_b,qkv_w,rpb  -> canonical slots:
    static const int bmap[13] = {FC1B, FC1W, FC2B, FC2W, N1B, N1G, N2B, N2G,
                                 PROJB, PROJW, QKVB, QKVW, RPB};
    x = F(0); ape = F(2);
    int idx = 3;
    for (int i = 0; i < 4; i++) for (int j = 0; j < 13; j++) db[i].p[bmap[j]] = F(idx++);
    for (int i = 0; i < 2; i++) for (int j = 0; j < 13; j++) eb[i].p[bmap[j]] = F(idx++);
    head_b = F(idx++); head_w = F(idx++);
    m4_nb = F(idx++); m4_ng = F(idx++); m4_w = F(idx++);
    m6_nb = F(idx++); m6_ng = F(idx++); m6_w = F(idx++);
    pe_b = F(idx++); pe_nb = F(idx++); pe_ng = F(idx++); pe_w = F(idx++);
    px_nb = F(idx++); px_ng = F(idx++); px_w = F(idx++);
    for (int i = 0; i < 4; i++) for (int j = 0; j < 13; j++) ub[i].p[bmap[j]] = F(idx++);
  }

  // workspace layout (floats). Xpe aliases tmp+qkv (+start of obuf); it is
  // only live before the first GEMM writes h. All big buffers have slack for
  // the padded M=2520->2528 GEMMs.
  float* W = (float*)d_ws;
  float* h      = W;                 // 60480*128
  float* tmp    = W + 7741440;       // 60480*128 (also 10080*768 / 2520*512 views)
  float* qkvB   = W + 15482880;      // 60480*384
  float* obuf   = W + 38707200;      // 60480*128
  float* xskip  = W + 46448640;      // 10080*128
  float* pe_wp  = W + 47738880;      // 128*544 zero-padded pe_w
  float* Xpe    = tmp;               // 60480*544 staging (aliased)

  auto gemm = [&](const float* A, int lda, const float* Wm, int ldw,
                  const float* bias, const float* res, float* Y,
                  int M, int N, int K, int act) {
    int Mtiles = (M + 15) / 16;
    long tiles = (long)Mtiles * (N / 64);
    int blocks = (int)((tiles + 7) / 8);
    gemm_bf16_wmma<<<blocks, 256, 0, stream>>>(A, Wm, bias, res, Y,
                                               Mtiles, N, K, lda, ldw, act);
  };
  auto ln = [&](const float* X, float* Y, const float* g, const float* b,
                int C, int M, const float* addB, int Lmod, const float* addR) {
    ln_kernel<<<M, 128, 0, stream>>>(X, Y, g, b, C, addB, Lmod, addR);
  };
  auto attn = [&](int WS, int shift, int L, int M, const float* rpb) {
    int nWtot = M / WS;
    if (WS == 6) {
      int blocks = (nWtot + 7) / 8;
      if (shift) attn_kernel<6, 1><<<blocks, 256, 0, stream>>>(qkvB, obuf, rpb, L, shift, nWtot);
      else       attn_kernel<6, 0><<<blocks, 256, 0, stream>>>(qkvB, obuf, rpb, L, shift, nWtot);
    } else if (WS == 35) {
      if (shift) attn_kernel<35, 1><<<nWtot, 256, 0, stream>>>(qkvB, obuf, rpb, L, shift, nWtot);
      else       attn_kernel<35, 0><<<nWtot, 256, 0, stream>>>(qkvB, obuf, rpb, L, shift, nWtot);
    } else {
      if (shift) attn_kernel<45, 1><<<nWtot, 256, 0, stream>>>(qkvB, obuf, rpb, L, shift, nWtot);
      else       attn_kernel<45, 0><<<nWtot, 256, 0, stream>>>(qkvB, obuf, rpb, L, shift, nWtot);
    }
  };
  auto swin = [&](const Blk& bp, int L, int WS, int shift, int M) {
    ln(h, tmp, bp.p[N1G], bp.p[N1B], 128, M, nullptr, 1, nullptr);
    gemm(tmp, 128, bp.p[QKVW], 128, bp.p[QKVB], nullptr, qkvB, M, 384, 128, 0);
    attn(WS, shift, L, M, bp.p[RPB]);
    gemm(obuf, 128, bp.p[PROJW], 128, bp.p[PROJB], h, h, M, 128, 128, 0);
    ln(h, tmp, bp.p[N2G], bp.p[N2B], 128, M, nullptr, 1, nullptr);
    gemm(tmp, 128, bp.p[FC1W], 128, bp.p[FC1B], nullptr, obuf, M, 128, 128, 1);
    gemm(obuf, 128, bp.p[FC2W], 128, bp.p[FC2B], h, h, M, 128, 128, 0);
  };

  // ---- patch embed: stage (60480 x 544) + padded weights, WMMA GEMM, LN ----
  pad_pew_kernel<<<(128 * 544 + 255) / 256, 256, 0, stream>>>(pe_w, pe_wp);
  const long totpe = 60480L * 544;
  stage_pe_kernel<<<(int)((totpe + 255) / 256), 256, 0, stream>>>(x, Xpe, totpe);
  gemm(Xpe, 544, pe_wp, 544, pe_b, nullptr, h, 60480, 128, 544, 0);
  ln(h, h, pe_ng, pe_nb, 128, 60480, ape, 7560, nullptr);

  // ---- epoch stage: 2 blocks, ws=6, L=7560 ----
  swin(eb[0], 7560, 6, 0, 60480);
  swin(eb[1], 7560, 6, 3, 60480);

  // ---- merge x6: LN(768) + GEMM -> x_skip; copy to h ----
  ln(h, tmp, m6_ng, m6_nb, 768, 10080, nullptr, 1, nullptr);
  gemm(tmp, 768, m6_w, 768, nullptr, nullptr, xskip, 10080, 128, 768, 0);
  copy_kernel<<<(int)((1290240 + 255) / 256), 256, 0, stream>>>(xskip, h, 1290240);

  // ---- down stage: 4 blocks, ws=35, L=1260 ----
  swin(db[0], 1260, 35, 0,  10080);
  swin(db[1], 1260, 35, 17, 10080);
  swin(db[2], 1260, 35, 0,  10080);
  swin(db[3], 1260, 35, 17, 10080);

  // ---- merge x4: LN(512) + GEMM ----
  ln(h, tmp, m4_ng, m4_nb, 512, 2520, nullptr, 1, nullptr);
  gemm(tmp, 512, m4_w, 512, nullptr, nullptr, h, 2520, 128, 512, 0);

  // ---- up stage: 4 blocks, ws=45, L=315 (M padded to 2528 inside gemm) ----
  swin(ub[0], 315, 45, 0,  2520);
  swin(ub[1], 315, 45, 22, 2520);
  swin(ub[2], 315, 45, 0,  2520);
  swin(ub[3], 315, 45, 22, 2520);

  // ---- patch expand: GEMM to 4C, reshape view, LN + x_skip residual ----
  gemm(h, 128, px_w, 128, nullptr, nullptr, tmp, 2520, 512, 128, 0);
  ln(tmp, h, px_ng, px_nb, 128, 10080, nullptr, 1, xskip);

  // ---- head ----
  head_kernel<<<(int)((10080 * 6 + 255) / 256), 256, 0, stream>>>(
      h, head_w, head_b, (float*)d_out, 10080);
}
